// MambaBlock_84181359002077
// MI455X (gfx1250) — matrix operand order repaired and checked by reading, not compiled, not run on hardware
//
#include <hip/hip_runtime.h>
#include <hip/hip_bf16.h>
#include <math.h>

// ---------------------------------------------------------------------------
// Mamba block for MI455X (gfx1250, wave32, WMMA), round 4.
// GEMM operands repacked once per call into fragment-contiguous f16 tiles
// (ISA 7.12.2 layouts).  WMMA inner loop is ping-pong unrolled by 2: the two
// fragment register sets alternate statically (no v_mov rotation), and loads
// for step k+1 overlap the v_wmma_f32_16x16x32_f16 issue for step k.
// Requires Kt even >= 2 (true for all call sites: 32, 64, 2, 64).
// ---------------------------------------------------------------------------

typedef __attribute__((ext_vector_type(16))) _Float16 v16h;
typedef __attribute__((ext_vector_type(8)))  float    v8f;

#define MB_B      2
#define MB_L      1024
#define MB_D      1024
#define MB_E      2048
#define MB_N      16
#define MB_KCONV  4
#define MB_R      64
#define MB_M      (MB_B * MB_L)     // 2048
#define MB_XDBL   (MB_R + 2*MB_N)   // 96

// ---------------------------------------------------------------------------
// Packed-fragment layouts (wave32, per CDNA5 ISA 7.12.2):
//  A tile (mt,kt): 32 lanes x 16 halves; lane l: m=l&15, g=l>>4;
//                  elem i<8 -> k=32kt+8g+i ; elem i>=8 -> k=32kt+8g+16+(i-8)
//  B tile (kt,nt): 32 lanes x 16 halves; lane l: n=16nt+(l&15), g=l>>4;
//                  elem i<8 -> k=32kt+8g+i ; elem i>=8 -> k=32kt+8g+16+(i-8)
// Fragment for (tile,lane) is 16 contiguous halves (32B, aligned).
// ---------------------------------------------------------------------------

__device__ __forceinline__ size_t packA_off(int row, int k, int Kt) {
    int mt = row >> 4, m = row & 15;
    int kt = k >> 5,  rem = k & 31;
    int g = (rem >> 3) & 1;                       // 0,1,0,1 per 8-chunk
    int i = (rem & 7) + ((rem & 16) ? 8 : 0);
    int lane = m + (g << 4);
    return ((size_t)(mt * Kt + kt) * 32 + lane) * 16 + i;
}

// ---- Repack f32 row-major [16*Mt, lda] -> packed-A f16 tiles ---------------
__global__ void pack_a_kernel(const float* __restrict__ src, _Float16* __restrict__ dst,
                              int Kt, int lda, int total /* Mt*Kt*32 */) {
    int t = blockIdx.x * blockDim.x + threadIdx.x;
    if (t >= total) return;
    int lane = t & 31, tile = t >> 5;
    int kt = tile % Kt, mt = tile / Kt;
    int m = lane & 15, g = lane >> 4;
    const float* sp = src + (size_t)((mt << 4) + m) * lda + (kt << 5) + (g << 3);
    _Float16* dp = dst + (size_t)t * 16;
#pragma unroll
    for (int i = 0; i < 8; ++i) {
        dp[i]     = (_Float16)sp[i];          // k = 32kt + 8g + i
        dp[i + 8] = (_Float16)sp[i + 16];     // k = 32kt + 8g + 16 + i
    }
}

// ---- Repack f32 row-major [32*Kt, ldb] -> packed-B f16 tiles ---------------
__global__ void pack_b_kernel(const float* __restrict__ src, _Float16* __restrict__ dst,
                              int Nt, int ldb, int total /* Kt*Nt*32 */) {
    int t = blockIdx.x * blockDim.x + threadIdx.x;
    if (t >= total) return;
    int lane = t & 31, tile = t >> 5;
    int nt = tile % Nt, kt = tile / Nt;
    int g = lane >> 4, n = (nt << 4) + (lane & 15);
    const float* sp = src + (size_t)((kt << 5) + (g << 3)) * ldb + n;
    _Float16* dp = dst + (size_t)t * 16;
#pragma unroll
    for (int i = 0; i < 16; ++i)
        dp[i] = (_Float16)sp[(size_t)(i + (i & 8)) * ldb];  // k = 32kt + 8g + i + (i&8)
}

// ---- Packed WMMA GEMM: C[16*Mt, N] = Apk @ Bpk  (f32 out) ------------------
// Wave computes a 16 x (16*WN) strip; block = 4 waves; grid.y = Mt.
// Ping-pong unroll-by-2 over kt (Kt must be even >= 2).
// EPI==1: C = softplus(C + bias[col]).
template <int WN, int EPI>
__global__ __launch_bounds__(128)
void wmma_gemm_packed(const _Float16* __restrict__ Apk, const _Float16* __restrict__ Bpk,
                      float* __restrict__ C, int Kt, int Nt, int N, int ldc,
                      const float* __restrict__ bias) {
    const int lane = threadIdx.x & 31;
    const int wave = threadIdx.x >> 5;
    const int nt0  = (blockIdx.x * (blockDim.x >> 5) + wave) * WN;
    const int col0 = nt0 << 4;
    const int mt   = blockIdx.y;
    if (col0 >= N) return;

    const v16h* __restrict__ Av = (const v16h*)Apk;
    const v16h* __restrict__ Bv = (const v16h*)Bpk;
    const v16h* ap = Av + (size_t)mt * Kt * 32 + lane;         // +32 per kt
    const v16h* bp = Bv + (size_t)nt0 * 32 + lane;             // +Nt*32 per kt, +32 per j
    const size_t bstride = (size_t)Nt * 32;

    v8f acc[WN];
#pragma unroll
    for (int j = 0; j < WN; ++j)
#pragma unroll
        for (int r = 0; r < 8; ++r) acc[j][r] = 0.0f;

    // prologue: set0 <- K-step 0
    v16h a0 = ap[0], a1;
    v16h b0[WN], b1[WN];
#pragma unroll
    for (int j = 0; j < WN; ++j) b0[j] = bp[(size_t)j * 32];

    int kt = 0;
    for (; kt + 2 < Kt; kt += 2) {
        // set1 <- kt+1, compute set0
        a1 = ap[(size_t)(kt + 1) * 32];
#pragma unroll
        for (int j = 0; j < WN; ++j)
            b1[j] = bp[(size_t)(kt + 1) * bstride + (size_t)j * 32];
#pragma unroll
        for (int j = 0; j < WN; ++j)
            acc[j] = __builtin_amdgcn_wmma_f32_16x16x32_f16(
                false, a0, false, b0[j], (short)0, acc[j], false, false);
        // set0 <- kt+2, compute set1
        a0 = ap[(size_t)(kt + 2) * 32];
#pragma unroll
        for (int j = 0; j < WN; ++j)
            b0[j] = bp[(size_t)(kt + 2) * bstride + (size_t)j * 32];
#pragma unroll
        for (int j = 0; j < WN; ++j)
            acc[j] = __builtin_amdgcn_wmma_f32_16x16x32_f16(
                false, a1, false, b1[j], (short)0, acc[j], false, false);
    }
    // tail pair: kt == Kt-2
    a1 = ap[(size_t)(kt + 1) * 32];
#pragma unroll
    for (int j = 0; j < WN; ++j)
        b1[j] = bp[(size_t)(kt + 1) * bstride + (size_t)j * 32];
#pragma unroll
    for (int j = 0; j < WN; ++j)
        acc[j] = __builtin_amdgcn_wmma_f32_16x16x32_f16(
            false, a0, false, b0[j], (short)0, acc[j], false, false);
#pragma unroll
    for (int j = 0; j < WN; ++j)
        acc[j] = __builtin_amdgcn_wmma_f32_16x16x32_f16(
            false, a1, false, b1[j], (short)0, acc[j], false, false);

    // C layout: VGPR r -> row = 16*mt + r + 8*(lane/16), col = col0+16j+(lane%16)
    const int n_lane  = lane & 15;
    const int row_off = (lane >> 4) * 8;
#pragma unroll
    for (int j = 0; j < WN; ++j) {
        const int col = col0 + 16 * j + n_lane;
        float* cp = C + (size_t)((mt << 4) + row_off) * ldc + col;
        float bcol = (EPI == 1) ? bias[col] : 0.0f;
#pragma unroll
        for (int r = 0; r < 8; ++r) {
            float v = acc[j][r];
            if (EPI == 1) {
                v += bcol;
                v = (v > 20.0f) ? v : log1pf(__expf(v));   // softplus
            }
            cp[(size_t)r * ldc] = v;
        }
    }
}

// ---- Causal depthwise conv (K=4) + SiLU; writes f32 u AND packed-A f16 -----
__global__ void conv_silu_pack_kernel(const float* __restrict__ xz,
                                      const float* __restrict__ conv_w,
                                      const float* __restrict__ conv_b,
                                      float* __restrict__ u,
                                      _Float16* __restrict__ uApk) {
    size_t i = (size_t)blockIdx.x * blockDim.x + threadIdx.x;   // over B*L*E
    int e = (int)(i % MB_E);
    size_t bt = i / MB_E;
    int t = (int)(bt % MB_L);
    size_t brow = bt - t;                                        // b*L
    float acc = conv_b[e];
#pragma unroll
    for (int k = 0; k < MB_KCONV; ++k) {
        int ts = t - (MB_KCONV - 1) + k;
        if (ts >= 0)
            acc += xz[(brow + ts) * (size_t)(2 * MB_E) + e] * conv_w[e * MB_KCONV + k];
    }
    float s = acc / (1.0f + __expf(-acc));                       // silu
    u[i] = s;
    uApk[packA_off((int)bt, e, MB_E / 32)] = (_Float16)s;
}

// ---- Selective scan (y overwrites u in place) ------------------------------
#define SCAN_CHUNK 64
__global__ __launch_bounds__(256)
void scan_kernel(const float* __restrict__ u_in,
                 const float* __restrict__ delta,
                 const float* __restrict__ A_log,
                 const float* __restrict__ xdbl,
                 const float* __restrict__ D_skip,
                 float* __restrict__ y_out) {
    const int c = blockIdx.x * blockDim.x + threadIdx.x;         // 0..B*E-1
    const int b = c / MB_E;
    const int e = c % MB_E;

    float a[MB_N], h[MB_N];
#pragma unroll
    for (int n = 0; n < MB_N; ++n) {
        a[n] = -__expf(A_log[(size_t)e * MB_N + n]);
        h[n] = 0.0f;
    }
    const float dsk = D_skip[e];

    __shared__ float sB[SCAN_CHUNK][MB_N];
    __shared__ float sC[SCAN_CHUNK][MB_N];
    const float* bc = xdbl + (size_t)b * MB_L * MB_XDBL + MB_R;  // cols 64..95

    for (int t0 = 0; t0 < MB_L; t0 += SCAN_CHUNK) {
        __syncthreads();
        for (int i = threadIdx.x; i < SCAN_CHUNK * 2 * MB_N; i += 256) {
            int tt = i >> 5, j = i & 31;
            float v = bc[(size_t)(t0 + tt) * MB_XDBL + j];
            if (j < MB_N) sB[tt][j] = v; else sC[tt][j - MB_N] = v;
        }
        __syncthreads();
        for (int tt = 0; tt < SCAN_CHUNK; ++tt) {
            const int t = t0 + tt;
            const size_t idx = ((size_t)b * MB_L + t) * MB_E + e;
            const float dlt = delta[idx];
            const float uv  = u_in[idx];
            const float du  = dlt * uv;
            float yv = 0.0f;
#pragma unroll
            for (int n = 0; n < MB_N; ++n) {
                float dA = __expf(dlt * a[n]);
                h[n] = dA * h[n] + du * sB[tt][n];
                yv += h[n] * sC[tt][n];
            }
            y_out[idx] = yv + uv * dsk;                          // in-place ok
        }
    }
}

// ---- Gate: write packed-A f16 of y*silu(z) directly ------------------------
__global__ void gate_pack_kernel(const float* __restrict__ y,
                                 const float* __restrict__ xz,
                                 _Float16* __restrict__ yApk) {
    size_t i = (size_t)blockIdx.x * blockDim.x + threadIdx.x;    // over B*L*E
    int e = (int)(i % MB_E);
    size_t bt = i / MB_E;
    float z = xz[bt * (size_t)(2 * MB_E) + MB_E + e];
    float g = y[i] * (z / (1.0f + __expf(-z)));
    yApk[packA_off((int)bt, e, MB_E / 32)] = (_Float16)g;
}

// ---------------------------------------------------------------------------

extern "C" void kernel_launch(void* const* d_in, const int* in_sizes, int n_in,
                              void* d_out, int out_size, void* d_ws, size_t ws_size,
                              hipStream_t stream) {
    const float* x       = (const float*)d_in[0];
    const float* W_in    = (const float*)d_in[1];
    const float* conv_w  = (const float*)d_in[2];
    const float* conv_b  = (const float*)d_in[3];
    const float* W_xp    = (const float*)d_in[4];
    const float* W_dt    = (const float*)d_in[5];
    const float* dt_bias = (const float*)d_in[6];
    const float* A_log   = (const float*)d_in[7];
    const float* D_skip  = (const float*)d_in[8];
    const float* W_out   = (const float*)d_in[9];
    float* out = (float*)d_out;

    // ---- workspace layout ----
    float* ws    = (float*)d_ws;
    float* xz    = ws;                                   // 2048*4096
    float* ubuf  = xz   + (size_t)MB_M * 2 * MB_E;       // 2048*2048 (u -> y)
    float* xdbl  = ubuf + (size_t)MB_M * MB_E;           // 2048*96
    float* delta = xdbl + (size_t)MB_M * MB_XDBL;        // 2048*2048
    _Float16* hb     = (_Float16*)(delta + (size_t)MB_M * MB_E);
    _Float16* WinPk  = hb;                               // 1024*4096
    _Float16* WxpPk  = WinPk  + (size_t)MB_D * 2 * MB_E; // 2048*96
    _Float16* WdtPk  = WxpPk  + (size_t)MB_E * MB_XDBL;  // 64*2048
    _Float16* WoutPk = WdtPk  + (size_t)MB_R * MB_E;     // 2048*1024
    _Float16* xApk   = WoutPk + (size_t)MB_E * MB_D;     // 2048*1024
    _Float16* uApk   = xApk   + (size_t)MB_M * MB_D;     // 2048*2048 (also yApk)
    _Float16* dtApk  = uApk   + (size_t)MB_M * MB_E;     // 2048*64

    const size_t elems_BLE = (size_t)MB_M * MB_E;

    // ---- 0) repack weights + x into WMMA fragment tiles ----
    { int tot = (MB_D/32)*(2*MB_E/16)*32;                 // W_in: Kt=32, Nt=256
      pack_b_kernel<<<(tot+255)/256, 256, 0, stream>>>(W_in, WinPk, 2*MB_E/16, 2*MB_E, tot); }
    { int tot = (MB_E/32)*(MB_XDBL/16)*32;                // W_xp: Kt=64, Nt=6
      pack_b_kernel<<<(tot+255)/256, 256, 0, stream>>>(W_xp, WxpPk, MB_XDBL/16, MB_XDBL, tot); }
    { int tot = (MB_R/32)*(MB_E/16)*32;                   // W_dt: Kt=2, Nt=128
      pack_b_kernel<<<(tot+255)/256, 256, 0, stream>>>(W_dt, WdtPk, MB_E/16, MB_E, tot); }
    { int tot = (MB_E/32)*(MB_D/16)*32;                   // W_out: Kt=64, Nt=64
      pack_b_kernel<<<(tot+255)/256, 256, 0, stream>>>(W_out, WoutPk, MB_D/16, MB_D, tot); }
    { int tot = (MB_M/16)*(MB_D/32)*32;                   // x: Mt=128, Kt=32
      pack_a_kernel<<<(tot+255)/256, 256, 0, stream>>>(x, xApk, MB_D/32, MB_D, tot); }

    // ---- 1) xz = x @ W_in   [2048x1024]@[1024x4096] ----
    wmma_gemm_packed<4, 0><<<dim3((2*MB_E)/256, MB_M/16), 128, 0, stream>>>(
        xApk, WinPk, xz, MB_D/32, 2*MB_E/16, 2*MB_E, 2*MB_E, nullptr);

    // ---- 2) causal conv + silu -> u (f32) + packed-A f16 ----
    conv_silu_pack_kernel<<<(unsigned)(elems_BLE/256), 256, 0, stream>>>(
        xz, conv_w, conv_b, ubuf, uApk);

    // ---- 3) x_dbl = u @ W_xp   [2048x2048]@[2048x96] ----
    wmma_gemm_packed<2, 0><<<dim3(1, MB_M/16), 128, 0, stream>>>(
        uApk, WxpPk, xdbl, MB_E/32, MB_XDBL/16, MB_XDBL, MB_XDBL, nullptr);

    // ---- 3b) pack dt_r (xdbl cols 0..63) for the delta GEMM ----
    { int tot = (MB_M/16)*(MB_R/32)*32;                   // Mt=128, Kt=2
      pack_a_kernel<<<(tot+255)/256, 256, 0, stream>>>(xdbl, dtApk, MB_R/32, MB_XDBL, tot); }

    // ---- 4) delta = softplus(dt_r @ W_dt + dt_bias)   [2048x64]@[64x2048] ----
    wmma_gemm_packed<4, 1><<<dim3(MB_E/256, MB_M/16), 128, 0, stream>>>(
        dtApk, WdtPk, delta, MB_R/32, MB_E/16, MB_E, MB_E, dt_bias);

    // ---- 5) selective scan (y overwrites ubuf) ----
    scan_kernel<<<(MB_B * MB_E)/256, 256, 0, stream>>>(
        ubuf, delta, A_log, xdbl, D_skip, ubuf);

    // ---- 6) gate: yApk = pack(y * silu(z))  (reuses uApk tiles) ----
    gate_pack_kernel<<<(unsigned)(elems_BLE/256), 256, 0, stream>>>(ubuf, xz, uApk);

    // ---- 7) out = y @ W_out   [2048x2048]@[2048x1024] ----
    wmma_gemm_packed<4, 0><<<dim3(MB_D/256, MB_M/16), 128, 0, stream>>>(
        uApk, WoutPk, out, MB_E/32, MB_D/16, MB_D, MB_D, nullptr);
}
